// MultiHeadAttention_8693013807367
// MI455X (gfx1250) — compile-verified, matrix operand order checked by reading, not run on hardware
//
#include <hip/hip_runtime.h>
#include <hip/hip_bf16.h>

typedef __attribute__((ext_vector_type(16))) _Float16 v16h;
typedef __attribute__((ext_vector_type(8)))  _Float16 v8h;
typedef __attribute__((ext_vector_type(8)))  float    v8f;

#define BATCH 2
#define SEQ   2048
#define CIN   512
#define NHEAD 8
#define CHEAD 64
#define HC    512          // NHEAD*CHEAD
#define BS    4096         // BATCH*SEQ

// ---------------------------------------------------------------------------
// WMMA helpers
// ---------------------------------------------------------------------------
__device__ __forceinline__ v8f wmma_f16(v16h a, v16h b, v8f c) {
  // D = A(16x32 f16) * B(32x16 f16) + C(16x16 f32)
  return __builtin_amdgcn_wmma_f32_16x16x32_f16(
      false, a, false, b, (short)0, c, false, false);
}

// Load one 16x32 f16 fragment (A-layout; also serves B when the B source is
// stored n-major with k contiguous). `base` points at element (row0, k0),
// `ld` is the row stride in elements. Per ISA 16-bit A layout:
//   lanes 0-15 : row = lane,    K = 0..7 and 16..23
//   lanes 16-31: row = lane-16, K = 8..15 and 24..31
__device__ __forceinline__ v16h load_frag16(const _Float16* base, int ld) {
  const int lane = threadIdx.x & 31;
  const int row  = lane & 15;
  const int koff = (lane >> 4) << 3;        // 0 or 8
  const _Float16* p = base + row * ld + koff;
  v8h lo = *(const v8h*)(p);                // K = koff .. koff+7
  v8h hi = *(const v8h*)(p + 16);           // K = 16+koff .. 16+koff+7
  return __builtin_shufflevector(lo, hi, 0,1,2,3,4,5,6,7,
                                 8,9,10,11,12,13,14,15);
}

// ---------------------------------------------------------------------------
// CDNA5 async memory->LDS staging (ASYNCcnt path; no VGPR round-trip).
// Async loads complete in order, so "s_wait_asynccnt N" with N = #chunks just
// issued for the NEXT tile guarantees the PREVIOUS tile has fully landed --
// the basis for double-buffered DMA/compute overlap.
// ---------------------------------------------------------------------------
typedef __attribute__((address_space(3))) const void* lds_cptr_t;

__device__ __forceinline__ unsigned lds_base(const void* p) {
  // addrspace(3) pointers are 32-bit offsets from the wave's LDS base
  return (unsigned)(unsigned long long)(lds_cptr_t)p;
}

__device__ __forceinline__ void async_ld16(unsigned lds_off, const void* g) {
  // per-lane: LDS[lds_off] = MEM[g] (16 bytes), tracked with ASYNCcnt
  asm volatile("global_load_async_to_lds_b128 %0, %1, off"
               :: "v"(lds_off), "v"((unsigned long long)g)
               : "memory");
}

__device__ __forceinline__ void wait_async0() {
  asm volatile("s_wait_asynccnt 0x0" ::: "memory");
}
__device__ __forceinline__ void wait_async2() {
  asm volatile("s_wait_asynccnt 0x2" ::: "memory");
}
__device__ __forceinline__ void wait_async4() {
  asm volatile("s_wait_asynccnt 0x4" ::: "memory");
}

// ---------------------------------------------------------------------------
// Conversion kernels
// ---------------------------------------------------------------------------
__global__ void __launch_bounds__(256) cvt_f32_to_f16(
    const float* __restrict__ src, _Float16* __restrict__ dst, int n) {
  int i = blockIdx.x * 256 + threadIdx.x;
  if (i < n) dst[i] = (_Float16)src[i];
}

// wT[n*512 + k] = (f16) w[k*512 + n]   (512x512 transpose + convert)
__global__ void __launch_bounds__(256) cvt_w_transpose(
    const float* __restrict__ w, _Float16* __restrict__ wT) {
  int i = blockIdx.x * 256 + threadIdx.x;   // 0 .. 262143
  int n = i >> 9, k = i & 511;
  wT[i] = (_Float16)w[k * CIN + n];
}

// ---------------------------------------------------------------------------
// Projection GEMM: Y[4096,512] = Xh[4096,512] @ W[512,512]  (W given as W^T)
// mode 0/1: Q/K  -> f16 [B,H,S,c];  mode 2: V -> f16 [B,H,c,S] (transposed);
// mode 3: gate -> f32 sigmoid(Y + bg) at [B,S,HC].
// Block: 128 threads = 4 waves; block tile 64(M) x 64(N); wave tile 16x64.
// W^T tiles double-buffered in LDS via async DMA, shared by the 4 waves.
// ---------------------------------------------------------------------------
__global__ void __launch_bounds__(128) gemm_proj(
    const _Float16* __restrict__ xh, const _Float16* __restrict__ wT,
    const float* __restrict__ bg,
    _Float16* __restrict__ out16, float* __restrict__ out32, int mode) {
  __shared__ _Float16 wbufA[64 * 32];       // [n_local][k_local]
  __shared__ _Float16 wbufB[64 * 32];

  const int tid  = threadIdx.x;
  const int wv   = tid >> 5;
  const int lane = tid & 31;
  const int lr   = lane & 15;
  const int lh   = lane >> 4;
  const int m0   = blockIdx.y * 64 + wv * 16;
  const int n0   = blockIdx.x * 64;
  const unsigned wA = lds_base(wbufA);
  const unsigned wB = lds_base(wbufB);

  v8f acc[4] = {};

  auto stage = [&](int k0, unsigned dst) {
#pragma unroll
    for (int i = 0; i < 2; ++i) {
      const int id = tid + i * 128;         // 0..255 chunks of 16B
      const int nl = id >> 2, ch = id & 3;
      async_ld16(dst + (unsigned)(nl * 32 + ch * 8) * 2,
                 wT + (size_t)(n0 + nl) * CIN + k0 + ch * 8);
    }
  };
  auto compute = [&](int k0, const _Float16* wb) {
    v16h a = load_frag16(xh + (size_t)m0 * CIN + k0, CIN);
#pragma unroll
    for (int nt = 0; nt < 4; ++nt) {
      v16h b = load_frag16(wb + nt * 16 * 32, 32);
      acc[nt] = wmma_f16(a, b, acc[nt]);
    }
  };

  stage(0, wA);
  for (int k0 = 0; k0 < CIN; k0 += 64) {
    stage(k0 + 32, wB);                     // k0+32 < 512 always
    wait_async2();                          // tile k0 landed; k0+32 in flight
    __syncthreads();
    compute(k0, wbufA);
    __syncthreads();                        // all reads of A done
    if (k0 + 64 < CIN) { stage(k0 + 64, wA); wait_async2(); }
    else               { wait_async0(); }
    __syncthreads();
    compute(k0 + 32, wbufB);
    __syncthreads();                        // all reads of B done
  }

  const int mbase = m0 + lh * 8;
#pragma unroll
  for (int nt = 0; nt < 4; ++nt) {
    const int n = n0 + nt * 16 + lr;
    const int h = n >> 6, c = n & 63;
#pragma unroll
    for (int r = 0; r < 8; ++r) {
      const int m = mbase + r;
      const int b = m >> 11, s = m & (SEQ - 1);
      const float v = acc[nt][r];
      if (mode <= 1) {
        out16[(((size_t)(b * NHEAD + h)) * SEQ + s) * CHEAD + c] = (_Float16)v;
      } else if (mode == 2) {
        out16[(((size_t)(b * NHEAD + h)) * CHEAD + c) * SEQ + s] = (_Float16)v;
      } else {
        out32[((size_t)(b * SEQ + s)) * HC + n] =
            1.0f / (1.0f + __expf(-(v + bg[n])));
      }
    }
  }
}

// ---------------------------------------------------------------------------
// Flash attention with pair bias + key mask + sigmoid gating.
// grid = (SEQ/64, B*H); block = 128 (4 waves); wave owns one 16-query tile.
// Keys streamed in steps of 32; K (4 KB) and V (4 KB) tiles double-buffered
// in LDS via async DMA so the next tile streams during this tile's WMMAs.
// ---------------------------------------------------------------------------
__global__ void __launch_bounds__(128) attn_flash(
    const _Float16* __restrict__ qh, const _Float16* __restrict__ kh,
    const _Float16* __restrict__ vt, const float* __restrict__ bias,
    const int* __restrict__ amask, const float* __restrict__ gate,
    _Float16* __restrict__ ocat) {
  __shared__ _Float16 kbufA[32 * 64], kbufB[32 * 64];   // [key_local][c]
  __shared__ _Float16 vbufA[64 * 32], vbufB[64 * 32];   // [c][key_local]
  __shared__ _Float16 pbuf[4][16 * 32];                 // per-wave P bounce

  const int tid  = threadIdx.x;
  const int wv   = tid >> 5;
  const int lane = tid & 31;
  const int lr   = lane & 15;
  const int lh   = lane >> 4;
  const int bh   = blockIdx.y;              // 0..15  => (b,h)
  const int bb   = bh >> 3, h = bh & 7;
  const int q0   = (blockIdx.x * 4 + wv) * 16;

  const _Float16* Q = qh + (size_t)bh * SEQ * CHEAD;
  const _Float16* K = kh + (size_t)bh * SEQ * CHEAD;
  const _Float16* V = vt + (size_t)bh * CHEAD * SEQ;
  const float* biasb = bias + ((size_t)bh * SEQ + q0) * SEQ;
  const int*   mk    = amask + bb * SEQ;
  const unsigned kA = lds_base(kbufA), kB = lds_base(kbufB);
  const unsigned vA = lds_base(vbufA), vB = lds_base(vbufB);

  v16h aq0 = load_frag16(Q + (size_t)q0 * CHEAD + 0,  CHEAD);
  v16h aq1 = load_frag16(Q + (size_t)q0 * CHEAD + 32, CHEAD);

  float mrun[8], lrun[8];
#pragma unroll
  for (int r = 0; r < 8; ++r) { mrun[r] = -1e30f; lrun[r] = 0.0f; }
  v8f oacc[4] = {};

  auto stage = [&](int kt, unsigned kdst, unsigned vdst) {
#pragma unroll
    for (int i = 0; i < 2; ++i) {
      const int id = tid + i * 128;         // 0..255
      const int kr = id >> 3, kc = id & 7;  // K: [32 rows][8 chunks]
      async_ld16(kdst + (unsigned)(kr * 64 + kc * 8) * 2,
                 K + (size_t)(kt + kr) * CHEAD + kc * 8);
      const int vc = id >> 2, vch = id & 3; // V^T: [64 rows][4 chunks]
      async_ld16(vdst + (unsigned)(vc * 32 + vch * 8) * 2,
                 V + (size_t)vc * SEQ + kt + vch * 8);
    }
  };

  auto compute = [&](int kt, const _Float16* kb, const _Float16* vb) {
    // --- logits tile: S = Q K^T  (16 x 32 in two 16x16 f32 fragments) ---
    v8f sa[2] = {};
#pragma unroll
    for (int nh = 0; nh < 2; ++nh) {
      v16h b0 = load_frag16(kb + nh * 16 * 64 + 0,  64);
      sa[nh] = wmma_f16(aq0, b0, sa[nh]);
      v16h b1 = load_frag16(kb + nh * 16 * 64 + 32, 64);
      sa[nh] = wmma_f16(aq1, b1, sa[nh]);
    }
    // --- scale + pair bias + key mask (bias streamed once from HBM) ---
#pragma unroll
    for (int nh = 0; nh < 2; ++nh) {
      const int kcol  = kt + nh * 16 + lr;
      const float moff = (mk[kcol] > 0) ? 0.0f : -1e9f;
      const float* bp = biasb + (size_t)(lh * 8) * SEQ + kcol;
      __builtin_prefetch(bp + 32, 0, 0);    // next key step -> global_prefetch
#pragma unroll
      for (int r = 0; r < 8; ++r)
        sa[nh][r] = sa[nh][r] * 0.125f + bp[(size_t)r * SEQ] + moff;
    }
    // --- online softmax (row reductions across 16 lanes, wave32 shfl) ---
#pragma unroll
    for (int r = 0; r < 8; ++r) {
      float tm = fmaxf(sa[0][r], sa[1][r]);
#pragma unroll
      for (int off = 8; off; off >>= 1) tm = fmaxf(tm, __shfl_xor(tm, off, 16));
      const float mnew = fmaxf(mrun[r], tm);
      const float p0 = __expf(sa[0][r] - mnew);
      const float p1 = __expf(sa[1][r] - mnew);
      sa[0][r] = p0; sa[1][r] = p1;
      float ts = p0 + p1;
#pragma unroll
      for (int off = 8; off; off >>= 1) ts += __shfl_xor(ts, off, 16);
      const float alpha = __expf(mrun[r] - mnew);
      lrun[r] = lrun[r] * alpha + ts;
      mrun[r] = mnew;
#pragma unroll
      for (int ct = 0; ct < 4; ++ct) oacc[ct][r] *= alpha;
    }
    // --- re-stripe P: C-layout f32 regs -> row-major f16 LDS -> A frag ---
    _Float16* pb = pbuf[wv];
#pragma unroll
    for (int nh = 0; nh < 2; ++nh)
#pragma unroll
      for (int r = 0; r < 8; ++r)
        pb[(lh * 8 + r) * 32 + nh * 16 + lr] = (_Float16)sa[nh][r];
    asm volatile("s_wait_dscnt 0x0" ::: "memory");
    v16h ap = load_frag16(pb, 32);
    // --- O += P V  (V staged transposed: rows = output column c) ---
#pragma unroll
    for (int ct = 0; ct < 4; ++ct) {
      v16h bv = load_frag16(vb + ct * 16 * 32, 32);
      oacc[ct] = wmma_f16(ap, bv, oacc[ct]);
    }
  };

  stage(0, kA, vA);
  for (int kt = 0; kt < SEQ; kt += 64) {
    stage(kt + 32, kB, vB);                 // kt+32 < 2048 always
    wait_async4();                          // tile kt landed; kt+32 in flight
    __syncthreads();
    compute(kt, kbufA, vbufA);
    __syncthreads();
    if (kt + 64 < SEQ) { stage(kt + 64, kA, vA); wait_async4(); }
    else               { wait_async0(); }
    __syncthreads();
    compute(kt + 32, kbufB, vbufB);
    __syncthreads();
  }

  // --- epilogue: normalize, gate, store concat-head f16 [B,S,HC] ---
#pragma unroll
  for (int r = 0; r < 8; ++r) {
    const float rl = 1.0f / lrun[r];
    const int q = q0 + lh * 8 + r;
    const size_t rowoff = ((size_t)(bb * SEQ + q)) * HC + h * CHEAD;
#pragma unroll
    for (int ct = 0; ct < 4; ++ct) {
      const int c = ct * 16 + lr;
      const float g = gate[rowoff + c];
      ocat[rowoff + c] = (_Float16)(oacc[ct][r] * rl * g);
    }
  }
}

// ---------------------------------------------------------------------------
// Output GEMM: out[4096,512] = Ocat[4096,512] @ Wo + bo   (Wo given as Wo^T)
// Same double-buffered async LDS weight staging as gemm_proj.
// ---------------------------------------------------------------------------
__global__ void __launch_bounds__(128) gemm_out(
    const _Float16* __restrict__ oc, const _Float16* __restrict__ woT,
    const float* __restrict__ bo, float* __restrict__ out) {
  __shared__ _Float16 wbufA[64 * 32];
  __shared__ _Float16 wbufB[64 * 32];

  const int tid  = threadIdx.x;
  const int wv   = tid >> 5;
  const int lane = tid & 31;
  const int lr   = lane & 15;
  const int lh   = lane >> 4;
  const int m0   = blockIdx.y * 64 + wv * 16;
  const int n0   = blockIdx.x * 64;
  const unsigned wA = lds_base(wbufA);
  const unsigned wB = lds_base(wbufB);

  v8f acc[4] = {};

  auto stage = [&](int k0, unsigned dst) {
#pragma unroll
    for (int i = 0; i < 2; ++i) {
      const int id = tid + i * 128;
      const int nl = id >> 2, ch = id & 3;
      async_ld16(dst + (unsigned)(nl * 32 + ch * 8) * 2,
                 woT + (size_t)(n0 + nl) * HC + k0 + ch * 8);
    }
  };
  auto compute = [&](int k0, const _Float16* wb) {
    v16h a = load_frag16(oc + (size_t)m0 * HC + k0, HC);
#pragma unroll
    for (int nt = 0; nt < 4; ++nt) {
      v16h b = load_frag16(wb + nt * 16 * 32, 32);
      acc[nt] = wmma_f16(a, b, acc[nt]);
    }
  };

  stage(0, wA);
  for (int k0 = 0; k0 < HC; k0 += 64) {
    stage(k0 + 32, wB);
    wait_async2();
    __syncthreads();
    compute(k0, wbufA);
    __syncthreads();
    if (k0 + 64 < HC) { stage(k0 + 64, wA); wait_async2(); }
    else              { wait_async0(); }
    __syncthreads();
    compute(k0 + 32, wbufB);
    __syncthreads();
  }

  const int mbase = m0 + lh * 8;
#pragma unroll
  for (int nt = 0; nt < 4; ++nt) {
    const int n = n0 + nt * 16 + lr;
    const float bn = bo[n];
#pragma unroll
    for (int r = 0; r < 8; ++r)
      out[(size_t)(mbase + r) * CIN + n] = acc[nt][r] + bn;
  }
}

// ---------------------------------------------------------------------------
extern "C" void kernel_launch(void* const* d_in, const int* in_sizes, int n_in,
                              void* d_out, int out_size, void* d_ws,
                              size_t ws_size, hipStream_t stream) {
  const float* x     = (const float*)d_in[0];
  const float* bias  = (const float*)d_in[1];
  const int*   amask = (const int*)d_in[2];
  const float* Wq    = (const float*)d_in[3];
  const float* Wk    = (const float*)d_in[4];
  const float* Wv    = (const float*)d_in[5];
  const float* Wg    = (const float*)d_in[6];
  const float* bg    = (const float*)d_in[7];
  const float* Wo    = (const float*)d_in[8];
  const float* bo    = (const float*)d_in[9];
  float* out = (float*)d_out;

  char* ws = (char*)d_ws;
  size_t off = 0;
  auto carve = [&](size_t bytes) -> char* {
    char* p = ws + off;
    off += (bytes + 255) & ~(size_t)255;
    return p;
  };
  _Float16* xh  = (_Float16*)carve((size_t)BS * CIN * 2);
  _Float16* wtq = (_Float16*)carve((size_t)CIN * HC * 2);
  _Float16* wtk = (_Float16*)carve((size_t)CIN * HC * 2);
  _Float16* wtv = (_Float16*)carve((size_t)CIN * HC * 2);
  _Float16* wtg = (_Float16*)carve((size_t)CIN * HC * 2);
  _Float16* wto = (_Float16*)carve((size_t)HC * CIN * 2);
  _Float16* qh  = (_Float16*)carve((size_t)BATCH * NHEAD * SEQ * CHEAD * 2);
  _Float16* kh  = (_Float16*)carve((size_t)BATCH * NHEAD * SEQ * CHEAD * 2);
  _Float16* vt  = (_Float16*)carve((size_t)BATCH * NHEAD * SEQ * CHEAD * 2);
  float*    gat = (float*)carve((size_t)BS * HC * 4);
  _Float16* oc  = (_Float16*)carve((size_t)BS * HC * 2);

  // 1) convert x to f16; transpose+convert the 5 weight matrices
  {
    const int n = BS * CIN;
    cvt_f32_to_f16<<<(n + 255) / 256, 256, 0, stream>>>(x, xh, n);
    const int wblk = (CIN * HC) / 256;
    cvt_w_transpose<<<wblk, 256, 0, stream>>>(Wq, wtq);
    cvt_w_transpose<<<wblk, 256, 0, stream>>>(Wk, wtk);
    cvt_w_transpose<<<wblk, 256, 0, stream>>>(Wv, wtv);
    cvt_w_transpose<<<wblk, 256, 0, stream>>>(Wg, wtg);
    cvt_w_transpose<<<wblk, 256, 0, stream>>>(Wo, wto);
  }

  // 2) projections (Q, K, V^T, gate)
  {
    dim3 grid(HC / 64, BS / 64);
    gemm_proj<<<grid, 128, 0, stream>>>(xh, wtq, bg, qh, gat, 0);
    gemm_proj<<<grid, 128, 0, stream>>>(xh, wtk, bg, kh, gat, 1);
    gemm_proj<<<grid, 128, 0, stream>>>(xh, wtv, bg, vt, gat, 2);
    gemm_proj<<<grid, 128, 0, stream>>>(xh, wtg, bg, qh /*unused*/, gat, 3);
  }

  // 3) fused flash attention (bias streamed once, softmax online, gated)
  {
    dim3 grid(SEQ / 64, BATCH * NHEAD);
    attn_flash<<<grid, 128, 0, stream>>>(qh, kh, vt, bias, amask, gat, oc);
  }

  // 4) output projection + bias
  {
    dim3 grid(CIN / 64, BS / 64);
    gemm_out<<<grid, 128, 0, stream>>>(oc, wto, bo, out);
  }
}
